// TransformerBlock_19069654794400
// MI455X (gfx1250) — matrix operand order repaired and checked by reading, not compiled, not run on hardware
//
#include <hip/hip_runtime.h>
#include <hip/hip_bf16.h>

#define DEV static __device__ __forceinline__

typedef __attribute__((ext_vector_type(16))) __bf16 v16bf;
typedef __attribute__((ext_vector_type(8)))  __bf16 v8bf;
typedef __attribute__((ext_vector_type(8)))  float  v8f;

namespace cfg {
constexpr int B  = 2;
constexpr int S  = 2048;
constexpr int H  = 16;
constexpr int DH = 64;
constexpr int DM = 1024;
constexpr int DF = 4096;
constexpr int T  = B * S;   // 4096 tokens
}

// ---------- helpers ----------

DEV unsigned short f2bf(float f) {
  union { float f; unsigned u; } c{f};
  unsigned u = c.u + 0x7FFFu + ((c.u >> 16) & 1u);   // round-to-nearest-even
  return (unsigned short)(u >> 16);
}

DEV v8f wmma_bf16(v16bf a, v16bf b, v8f c) {
  // D = A(16x32 bf16) * B(32x16 bf16) + C(16x16 f32)
  return __builtin_amdgcn_wmma_f32_16x16x32_bf16(
      /*neg_a=*/false, a, /*neg_b=*/false, b,
      /*c_mod=*/(short)0, c, /*reuse_a=*/false, /*reuse_b=*/false);
}

// A-matrix 16x32 bf16 fragment: lane L<16 -> row L, K 0..15;
// lane L>=16 -> row L-16, K 16..31. base = &tile(row0, k0), stride in elems.
DEV v16bf frag_a(const unsigned short* base, int stride) {
  const int lane = threadIdx.x & 31;
  const int row  = lane & 15;
  const int ko   = (lane >> 4) * 16;
  const __bf16* p = (const __bf16*)(base + (size_t)row * stride + ko);
  union { v16bf v; v8bf h[2]; } r;
  r.h[0] = *(const v8bf*)p;
  r.h[1] = *(const v8bf*)(p + 8);
  return r.v;
}

// B-matrix 32x16 bf16 fragment: lane n%16 holds column n; lanes<16 K 0..15,
// lanes>=16 K 16..31. Source layout is B^T row-major: [N][K] -> contiguous 32B.
DEV v16bf frag_b(const unsigned short* base, int stride) {
  const int lane = threadIdx.x & 31;
  const int row  = lane & 15;           // column N
  const int ko   = (lane >> 4) * 16;    // K chunk
  return *(const v16bf*)(base + (size_t)row * stride + ko);
}

DEV float block_sum(float v, float* sh) {
#pragma unroll
  for (int m = 16; m; m >>= 1) v += __shfl_xor(v, m, 32);
  const int w = threadIdx.x >> 5, l = threadIdx.x & 31;
  if (l == 0) sh[w] = v;
  __syncthreads();
  float r = (l < 8) ? sh[l] : 0.f;
#pragma unroll
  for (int m = 4; m; m >>= 1) r += __shfl_xor(r, m, 32);
  r = __shfl(r, 0, 32);
  __syncthreads();
  return r;
}

// ---------- weight packing ----------

// W_Q/K/V: [H][DM][DH] fp32 -> Bp[n][k] bf16 with n = h*64+d, k = m
__global__ __launch_bounds__(256) void pack_qkv(const float* __restrict__ Wsrc,
                                                unsigned short* __restrict__ dst) {
  const size_t i = (size_t)blockIdx.x * 256 + threadIdx.x;   // over 1024*1024
  const int n = (int)(i >> 10), k = (int)(i & 1023);
  const int h = n >> 6, d = n & 63;
  dst[i] = f2bf(Wsrc[((size_t)h * 1024 + k) * 64 + d]);
}

// [K][N] fp32 row-major -> [N][K] bf16 (B^T)
__global__ __launch_bounds__(256) void pack_transpose(const float* __restrict__ Wsrc,
                                                      unsigned short* __restrict__ dst,
                                                      int K, int N) {
  const size_t i = (size_t)blockIdx.x * 256 + threadIdx.x;   // over N*K
  const int n = (int)(i / K), k = (int)(i % K);
  dst[i] = f2bf(Wsrc[(size_t)k * N + n]);
}

// ---------- layernorm (fp32 in -> bf16 out) ----------

__global__ __launch_bounds__(256) void layernorm_bf16(const float* __restrict__ x,
                                                      const float* __restrict__ w,
                                                      const float* __restrict__ b,
                                                      unsigned short* __restrict__ out) {
  __shared__ float sh[8];
  const int row = blockIdx.x;
  const int tid = threadIdx.x;
  const float* xr = x + (size_t)row * 1024;
  float v[4];
  float s = 0.f;
#pragma unroll
  for (int i = 0; i < 4; ++i) { v[i] = xr[tid + i * 256]; s += v[i]; }
  s = block_sum(s, sh);
  const float mean = s * (1.f / 1024.f);
  float q = 0.f;
#pragma unroll
  for (int i = 0; i < 4; ++i) { v[i] -= mean; q += v[i] * v[i]; }
  q = block_sum(q, sh);
  const float rstd = rsqrtf(q * (1.f / 1024.f) + 1e-5f);
#pragma unroll
  for (int i = 0; i < 4; ++i) {
    const int c = tid + i * 256;
    out[(size_t)row * 1024 + c] = f2bf(v[i] * rstd * w[c] + b[c]);
  }
}

// ---------- generic WMMA GEMM: C[M,N] = A[M,K](bf16) * Bp[N,K]^T(bf16) ----------

enum { EP_QK = 0, EP_VT, EP_RESID, EP_GELU, EP_FINAL };

template <int EP>
__global__ __launch_bounds__(256) void gemm_bf16(const unsigned short* __restrict__ A,
                                                 const unsigned short* __restrict__ Bp,
                                                 const float* __restrict__ bias,
                                                 const float* __restrict__ resid,
                                                 void* __restrict__ outv,
                                                 int M, int N, int K) {
  using namespace cfg;
  __shared__ unsigned short As[64 * 32];    // block M tile x K-step
  __shared__ unsigned short Bs[256 * 32];   // block N tile x K-step
  const int tid  = threadIdx.x;
  const int wave = tid >> 5, lane = tid & 31;
  const int wm = wave >> 2, wn = wave & 3;  // 2 x 4 wave grid
  const int n0 = blockIdx.x * 256;
  const int m0 = blockIdx.y * 64;

  const v8f vz = {0.f, 0.f, 0.f, 0.f, 0.f, 0.f, 0.f, 0.f};
  v8f acc[2][4];
#pragma unroll
  for (int i = 0; i < 2; ++i)
#pragma unroll
    for (int j = 0; j < 4; ++j) acc[i][j] = vz;

  for (int k0 = 0; k0 < K; k0 += 32) {
    {  // stage A: 64x32 = 256 uint4, one per thread
      const int row = tid >> 2, kc = (tid & 3) * 8;
      *(uint4*)&As[row * 32 + kc] =
          *(const uint4*)&A[(size_t)(m0 + row) * K + k0 + kc];
    }
#pragma unroll
    for (int i = 0; i < 4; ++i) {  // stage B: 256x32 = 1024 uint4
      const int idx = tid + i * 256, row = idx >> 2, kc = (idx & 3) * 8;
      *(uint4*)&Bs[row * 32 + kc] =
          *(const uint4*)&Bp[(size_t)(n0 + row) * K + k0 + kc];
    }
    __syncthreads();
    const v16bf af0 = frag_a(&As[(wm * 32 + 0) * 32], 32);
    const v16bf af1 = frag_a(&As[(wm * 32 + 16) * 32], 32);
#pragma unroll
    for (int nt = 0; nt < 4; ++nt) {
      const v16bf bfr = frag_b(&Bs[(wn * 64 + nt * 16) * 32], 32);
      acc[0][nt] = wmma_bf16(af0, bfr, acc[0][nt]);
      acc[1][nt] = wmma_bf16(af1, bfr, acc[1][nt]);
    }
    __syncthreads();
  }

  // epilogue: C layout lane<16 -> M=v, lane>=16 -> M=v+8; N = lane%16
  const int colL = lane & 15;
  const int rowH = (lane >> 4) * 8;
#pragma unroll
  for (int mt = 0; mt < 2; ++mt)
#pragma unroll
    for (int nt = 0; nt < 4; ++nt)
#pragma unroll
      for (int v = 0; v < 8; ++v) {
        const int gm = m0 + wm * 32 + mt * 16 + rowH + v;
        const int gn = n0 + wn * 64 + nt * 16 + colL;
        const float val = acc[mt][nt][v] + bias[gn];
        if constexpr (EP == EP_QK || EP == EP_VT) {
          const int bi = gm >> 11, s = gm & (S - 1);   // S == 2048
          const int h = gn >> 6, d = gn & 63;
          unsigned short* o = (unsigned short*)outv;
          const size_t idx = (EP == EP_QK)
              ? ((((size_t)bi * H + h) * S + s) * 64 + d)     // [B,H,S,64]
              : ((((size_t)bi * H + h) * 64 + d) * S + s);    // [B,H,64,S]
          o[idx] = f2bf(val);
        } else if constexpr (EP == EP_RESID || EP == EP_FINAL) {
          float* o = (float*)outv;
          o[(size_t)gm * N + gn] = val + resid[(size_t)gm * N + gn];
        } else {  // EP_GELU, exact gelu
          unsigned short* o = (unsigned short*)outv;
          const float g = 0.5f * val * (1.f + erff(val * 0.70710678118f));
          o[(size_t)gm * N + gn] = f2bf(g);
        }
      }
}

// ---------- flash attention (causal, online softmax, all-WMMA) ----------

__global__ __launch_bounds__(128) void attn_flash(const unsigned short* __restrict__ Q,
                                                  const unsigned short* __restrict__ Kc,
                                                  const unsigned short* __restrict__ Vt,
                                                  unsigned short* __restrict__ Z) {
  using namespace cfg;
  __shared__ unsigned short Pl[4][16 * 32];   // per-wave P tile (16q x 32k) bf16
  const int bh   = blockIdx.y;                // b*H + h
  const int wave = threadIdx.x >> 5, lane = threadIdx.x & 31;
  const int qb   = blockIdx.x * 64 + wave * 16;   // this wave's 16 query rows
  const int colL = lane & 15, lh = lane >> 4;
  const unsigned short* Qp = Q  + (size_t)bh * S * 64;
  const unsigned short* Kp = Kc + (size_t)bh * S * 64;
  const unsigned short* Vp = Vt + (size_t)bh * 64 * S;
  unsigned short* Pw = Pl[wave];

  // Q fragments (16 rows x 64 d), reused for every key tile
  const v16bf qf0 = frag_a(Qp + (size_t)qb * 64, 64);
  const v16bf qf1 = frag_a(Qp + (size_t)qb * 64 + 32, 64);

  const v8f vz = {0.f, 0.f, 0.f, 0.f, 0.f, 0.f, 0.f, 0.f};
  v8f accO[4];
#pragma unroll
  for (int f = 0; f < 4; ++f) accO[f] = vz;
  float mrow[8], lrow[8];
#pragma unroll
  for (int v = 0; v < 8; ++v) { mrow[v] = -3.0e38f; lrow[v] = 0.f; }

  for (int kb = 0; kb <= qb + 15; kb += 32) {   // causal key-tile loop
    v8f s0 = vz, s1 = vz;
    v16bf bk;
    bk = frag_b(Kp + (size_t)(kb +  0) * 64 +  0, 64); s0 = wmma_bf16(qf0, bk, s0);
    bk = frag_b(Kp + (size_t)(kb +  0) * 64 + 32, 64); s0 = wmma_bf16(qf1, bk, s0);
    bk = frag_b(Kp + (size_t)(kb + 16) * 64 +  0, 64); s1 = wmma_bf16(qf0, bk, s1);
    bk = frag_b(Kp + (size_t)(kb + 16) * 64 + 32, 64); s1 = wmma_bf16(qf1, bk, s1);

    float corr[8];
#pragma unroll
    for (int v = 0; v < 8; ++v) {
      const int qr  = qb + lh * 8 + v;
      const int k0i = kb + colL, k1i = kb + 16 + colL;
      const float a0 = (k0i <= qr) ? s0[v] * 0.125f : -1e5f;   // 1/sqrt(64)
      const float a1 = (k1i <= qr) ? s1[v] * 0.125f : -1e5f;
      float mx = fmaxf(a0, a1);
#pragma unroll
      for (int m = 8; m; m >>= 1) mx = fmaxf(mx, __shfl_xor(mx, m, 32));
      const float nm = fmaxf(mrow[v], mx);
      corr[v] = __expf(mrow[v] - nm);
      const float p0 = __expf(a0 - nm), p1 = __expf(a1 - nm);
      float ps = p0 + p1;
#pragma unroll
      for (int m = 8; m; m >>= 1) ps += __shfl_xor(ps, m, 32);
      lrow[v] = lrow[v] * corr[v] + ps;
      mrow[v] = nm;
      Pw[(lh * 8 + v) * 32 + colL]      = f2bf(p0);
      Pw[(lh * 8 + v) * 32 + 16 + colL] = f2bf(p1);
    }
#pragma unroll
    for (int f = 0; f < 4; ++f)
#pragma unroll
      for (int v = 0; v < 8; ++v) accO[f][v] = accO[f][v] * corr[v];

    asm volatile("" ::: "memory");   // keep P stores before the fragment re-read
    const v16bf pf = frag_a(Pw, 32); // P as A-matrix 16x32
#pragma unroll
    for (int nd = 0; nd < 4; ++nd) {
      const v16bf vf = frag_b(Vp + (size_t)(nd * 16) * S + kb, S);  // V^T [d][s]
      accO[nd] = wmma_bf16(pf, vf, accO[nd]);
    }
  }

  const int bi = bh / H, h = bh % H;
#pragma unroll
  for (int nd = 0; nd < 4; ++nd)
#pragma unroll
    for (int v = 0; v < 8; ++v) {
      const int qr = qb + lh * 8 + v;
      const float o = accO[nd][v] / lrow[v];
      Z[((size_t)bi * S + qr) * 1024 + h * 64 + nd * 16 + colL] = f2bf(o);
    }
}

// ---------- host-side launch ----------

extern "C" void kernel_launch(void* const* d_in, const int* in_sizes, int n_in,
                              void* d_out, int out_size, void* d_ws, size_t ws_size,
                              hipStream_t stream) {
  using namespace cfg;
  (void)in_sizes; (void)n_in; (void)out_size; (void)ws_size;

  const float* x    = (const float*)d_in[0];
  const float* ln1w = (const float*)d_in[1];
  const float* ln1b = (const float*)d_in[2];
  const float* WQ   = (const float*)d_in[3];
  const float* WK   = (const float*)d_in[4];
  const float* WV   = (const float*)d_in[5];
  const float* WO   = (const float*)d_in[6];
  const float* bQ   = (const float*)d_in[7];
  const float* bK   = (const float*)d_in[8];
  const float* bV   = (const float*)d_in[9];
  const float* bO   = (const float*)d_in[10];
  const float* ln2w = (const float*)d_in[11];
  const float* ln2b = (const float*)d_in[12];
  const float* Win  = (const float*)d_in[13];
  const float* bin  = (const float*)d_in[14];
  const float* Wout = (const float*)d_in[15];
  const float* bout = (const float*)d_in[16];

  char* ws = (char*)d_ws;
  size_t off = 0;
  auto alloc = [&](size_t bytes) -> void* {
    void* p = ws + off;
    off += (bytes + 255) & ~(size_t)255;
    return p;
  };
  unsigned short* xln1  = (unsigned short*)alloc((size_t)T * DM * 2);
  unsigned short* wqp   = (unsigned short*)alloc((size_t)DM * DM * 2);
  unsigned short* wkp   = (unsigned short*)alloc((size_t)DM * DM * 2);
  unsigned short* wvp   = (unsigned short*)alloc((size_t)DM * DM * 2);
  unsigned short* wop   = (unsigned short*)alloc((size_t)DM * DM * 2);
  unsigned short* winp  = (unsigned short*)alloc((size_t)DM * DF * 2);
  unsigned short* woutp = (unsigned short*)alloc((size_t)DF * DM * 2);
  unsigned short* qbuf  = (unsigned short*)alloc((size_t)T * DM * 2);
  unsigned short* kbuf  = (unsigned short*)alloc((size_t)T * DM * 2);
  unsigned short* vtb   = (unsigned short*)alloc((size_t)T * DM * 2);
  unsigned short* zbuf  = (unsigned short*)alloc((size_t)T * DM * 2);
  float*          r1    = (float*)alloc((size_t)T * DM * 4);
  unsigned short* xln2  = (unsigned short*)alloc((size_t)T * DM * 2);
  unsigned short* hact  = (unsigned short*)alloc((size_t)T * DF * 2);

  // 1) pack weights -> bf16 [N][K]
  pack_qkv<<<(DM * DM) / 256, 256, 0, stream>>>(WQ, wqp);
  pack_qkv<<<(DM * DM) / 256, 256, 0, stream>>>(WK, wkp);
  pack_qkv<<<(DM * DM) / 256, 256, 0, stream>>>(WV, wvp);
  pack_transpose<<<(DM * DM) / 256, 256, 0, stream>>>(WO,   wop,   DM, DM);
  pack_transpose<<<(DM * DF) / 256, 256, 0, stream>>>(Win,  winp,  DM, DF);
  pack_transpose<<<(DF * DM) / 256, 256, 0, stream>>>(Wout, woutp, DF, DM);

  // 2) ln1 -> bf16 activations
  layernorm_bf16<<<T, 256, 0, stream>>>(x, ln1w, ln1b, xln1);

  // 3) QKV projections (WMMA GEMMs with layout epilogues)
  const dim3 gDM(DM / 256, T / 64);
  gemm_bf16<EP_QK><<<gDM, 256, 0, stream>>>(xln1, wqp, bQ, nullptr, qbuf, T, DM, DM);
  gemm_bf16<EP_QK><<<gDM, 256, 0, stream>>>(xln1, wkp, bK, nullptr, kbuf, T, DM, DM);
  gemm_bf16<EP_VT><<<gDM, 256, 0, stream>>>(xln1, wvp, bV, nullptr, vtb,  T, DM, DM);

  // 4) causal flash attention -> Z [T, H*DH]
  attn_flash<<<dim3(S / 64, B * H), 128, 0, stream>>>(qbuf, kbuf, vtb, zbuf);

  // 5) output projection + residual (fp32)
  gemm_bf16<EP_RESID><<<gDM, 256, 0, stream>>>(zbuf, wop, bO, x, r1, T, DM, DM);

  // 6) ln2 -> bf16
  layernorm_bf16<<<T, 256, 0, stream>>>(r1, ln2w, ln2b, xln2);

  // 7) MLP up + exact GELU -> bf16
  gemm_bf16<EP_GELU><<<dim3(DF / 256, T / 64), 256, 0, stream>>>(
      xln2, winp, bin, nullptr, hact, T, DF, DM);

  // 8) MLP down + residual -> fp32 output
  gemm_bf16<EP_FINAL><<<gDM, 256, 0, stream>>>(
      hact, woutp, bout, r1, (float*)d_out, T, DM, DF);
}
